// HAB_15539191677366
// MI455X (gfx1250) — matrix operand order repaired and checked by reading, not compiled, not run on hardware
//
#include <hip/hip_runtime.h>
#include <hip/hip_bf16.h>

// ---------------------------------------------------------------------------
// HAT/Swin block forward for MI455X (gfx1250, wave32, WMMA 16x16x32 f16).
// Every GEMM-like op (qkv / QK^T / PV / proj / fc1 / fc2 AND both 3x3 convs
// via 9-tap implicit GEMM) runs on v_wmma_f32_16x16x32_f16 with f32 accum.
// Weights are stored pre-swizzled in WMMA B-fragment order so GEMM staging is
// pure contiguous b128 copies (no LDS transpose scatter).  Activations are
// L2-resident (192MB L2), so 9-tap conv re-reads hit L2, not HBM.
// ---------------------------------------------------------------------------

typedef __attribute__((ext_vector_type(16))) _Float16 v16h;
typedef __attribute__((ext_vector_type(8)))  float    v8f;

#define T_TOK   131072      // 2 * 256 * 256 tokens
#define DIM     180
#define HID     720
#define CDIM    60
#define NH      6
#define HD      30
#define WSZ     16
#define SHIFT   8
#define DIMP    192         // DIM padded to 32
#define CDIMP   64          // CDIM padded to 64
#define QKVP    576         // 3*192 packed qkv layout per token
#define HIDP    736         // HID padded to 32 (23*32)
#define NS_QKV  576         // qkv B matrix padded cols (9*64)
#define NS_FC1  768         // fc1 B matrix padded cols (12*64)

// A-fragment k-map used by this code (contiguous, same order as the B swizzle): lane<16: k=0..15; lane>=16: k=16..31
__device__ __forceinline__ int kmapA(int h, int hi) {
    return h + hi * 16;
}

__device__ __forceinline__ float gelu_exact(float v) {
    return 0.5f * v * (1.0f + erff(v * 0.70710678118654752f));
}

// shift-mask region id on the un-rolled grid: [0,240) / [240,248) / [248,256)
__device__ __forceinline__ int grp(int p) { return (p < 240) ? 0 : ((p < 248) ? 1 : 2); }

// ---------------------------------------------------------------------------
// Weight convert: W [N][K] fp32 -> swizzled WMMA-B fragment order f16:
//   out[(kb*NT + ntile)*512 + lane*16 + h] = W[n][k],
//   k = kb*32 + (lane>>4)*16 + h,  n = ntile*16 + (lane&15),  NT = Np/16.
// ---------------------------------------------------------------------------
__global__ void convert_wT(const float* __restrict__ W, _Float16* __restrict__ out,
                           int N, int K, int Np, int Kp) {
    long idx = (long)blockIdx.x * 256 + threadIdx.x;
    if (idx >= (long)Np * Kp) return;
    int NT = Np >> 4;
    int h    = (int)(idx & 15);
    int lane = (int)((idx >> 4) & 31);
    long tl  = idx >> 9;
    int ntile = (int)(tl % NT);
    int kb    = (int)(tl / NT);
    int k = kb * 32 + (lane >> 4) * 16 + h;
    int n = ntile * 16 + (lane & 15);
    out[idx] = (k < K && n < N) ? (_Float16)W[(long)n * K + k] : (_Float16)0.0f;
}

// Conv weight: W [CO][CI][3][3] fp32 -> tap-major swizzled B operand,
// logical B[ktot][co] with ktot = t9*CIp + ci, same fragment swizzle as above.
__global__ void convert_conv_w(const float* __restrict__ W, _Float16* __restrict__ out,
                               int CO, int CI, int CIp, int COp) {
    long idx = (long)blockIdx.x * 256 + threadIdx.x;
    if (idx >= (long)9 * CIp * COp) return;
    int NT = COp >> 4;
    int h    = (int)(idx & 15);
    int lane = (int)((idx >> 4) & 31);
    long tl  = idx >> 9;
    int ntile = (int)(tl % NT);
    int kb    = (int)(tl / NT);
    int ktot = kb * 32 + (lane >> 4) * 16 + h;
    int co   = ntile * 16 + (lane & 15);
    int t9 = ktot / CIp, ci = ktot % CIp;
    out[idx] = (ci < CI && co < CO) ? (_Float16)W[((long)co * CI + ci) * 9 + t9]
                                    : (_Float16)0.0f;
}

// Zero the 12 pad halves (cols 180..191) of a [T_TOK][192] f16 buffer.
__global__ void zero_pads(_Float16* __restrict__ buf) {
    long idx = (long)blockIdx.x * 256 + threadIdx.x;
    long t = idx / 12;
    int  j = (int)(idx % 12);
    buf[t * DIMP + DIM + j] = (_Float16)0.0f;
}

// ---------------------------------------------------------------------------
// LayerNorm over 180 channels. One wave32 per token (8 tokens / 256-thr block).
// ---------------------------------------------------------------------------
__global__ __launch_bounds__(256) void ln_kernel(const float* __restrict__ in,
                                                 const float* __restrict__ g,
                                                 const float* __restrict__ b,
                                                 _Float16* __restrict__ outh) {
    int wave = threadIdx.x >> 5, lane = threadIdx.x & 31;
    long t = (long)blockIdx.x * 8 + wave;
    const float* p = in + t * DIM;
    float s = 0.f, s2 = 0.f;
    for (int c = lane; c < DIM; c += 32) { float v = p[c]; s += v; s2 += v * v; }
#pragma unroll
    for (int m = 16; m; m >>= 1) { s += __shfl_xor(s, m, 32); s2 += __shfl_xor(s2, m, 32); }
    float mu  = s  * (1.0f / DIM);
    float var = s2 * (1.0f / DIM) - mu * mu;
    float inv = rsqrtf(var + 1e-5f);
    for (int c = lane; c < DIM; c += 32) {
        float v = (p[c] - mu) * inv * g[c] + b[c];
        outh[t * DIMP + c] = (_Float16)v;
    }
}

// Global average pool over 65536 pixels per (b, c).
__global__ __launch_bounds__(256) void pool_kernel(const float* __restrict__ y2,
                                                   float* __restrict__ pool) {
    int bc = blockIdx.x;             // 0..359
    int b = bc / DIM, c = bc % DIM;
    const float* p = y2 + (long)b * 65536 * DIM + c;
    float s = 0.f;
    for (int t = threadIdx.x; t < 65536; t += 256) s += p[(long)t * DIM];
    __shared__ float sm[256];
    sm[threadIdx.x] = s;
    __syncthreads();
    for (int o = 128; o; o >>= 1) {
        if (threadIdx.x < o) sm[threadIdx.x] += sm[threadIdx.x + o];
        __syncthreads();
    }
    if (threadIdx.x == 0) pool[bc] = sm[0] * (1.0f / 65536.0f);
}

// Channel attention: sigmoid(W2 @ relu(W1 @ pool + b1) + b2).  One block / batch.
__global__ void ca_kernel(const float* __restrict__ pool, const float* __restrict__ w1,
                          const float* __restrict__ b1, const float* __restrict__ w2,
                          const float* __restrict__ b2, float* __restrict__ ca_a) {
    __shared__ float ps[DIM];
    __shared__ float s6[6];
    int b = blockIdx.x, tid = threadIdx.x;
    if (tid < DIM) ps[tid] = pool[b * DIM + tid];
    __syncthreads();
    if (tid < 6) {
        float s = b1[tid];
        for (int c = 0; c < DIM; ++c) s += w1[tid * DIM + c] * ps[c];
        s6[tid] = fmaxf(s, 0.f);
    }
    __syncthreads();
    if (tid < DIM) {
        float s = b2[tid];
        for (int j = 0; j < 6; ++j) s += w2[tid * 6 + j] * s6[j];
        ca_a[b * DIM + tid] = 1.0f / (1.0f + expf(-s));
    }
}

// ---------------------------------------------------------------------------
// Shared WMMA GEMM core: block tile 64(M) x 64(N), K in steps of 32.
// 8 waves; wave computes 16x32 (two 16x16 C tiles).
// A [M][Ks] f16 row-major; B pre-swizzled fragment-linear (see convert_wT).
// smA stride 40 halves (80B, 16B-aligned rows); smB 2048 halves linear.
// ---------------------------------------------------------------------------
__device__ __forceinline__ void wmma_step(const _Float16* smA, const _Float16* smB,
                                          int mt, int ntB, int lane, int l, int hi,
                                          v8f& c0, v8f& c1) {
    v16h a, b0, b1;
    {
        const _Float16* p = smA + (mt * 16 + l) * 40;
#pragma unroll
        for (int h = 0; h < 16; ++h) a[h] = p[kmapA(h, hi)];
    }
    {
        const _Float16* p = smB + ntB * 512 + lane * 16;
#pragma unroll
        for (int h = 0; h < 16; ++h) b0[h] = p[h];
    }
    {
        const _Float16* p = smB + (ntB + 1) * 512 + lane * 16;
#pragma unroll
        for (int h = 0; h < 16; ++h) b1[h] = p[h];
    }
    c0 = __builtin_amdgcn_wmma_f32_16x16x32_f16(false, a, false, b0, (short)0, c0, false, false);
    c1 = __builtin_amdgcn_wmma_f32_16x16x32_f16(false, a, false, b1, (short)0, c1, false, false);
}

__device__ __forceinline__ void gemm_tiles(const _Float16* __restrict__ A,
                                           const _Float16* __restrict__ Bsw,
                                           long rowBase, int colBase, int Ks, int NT,
                                           _Float16* smA /*[64*40]*/,
                                           _Float16* smB /*[2048]*/,
                                           v8f& c0, v8f& c1) {
    int tid = threadIdx.x, wave = tid >> 5, lane = tid & 31;
    int l = lane & 15, hi = lane >> 4;
    int mt = wave & 3, ntB = (wave >> 2) * 2;
    int arow = tid >> 2, aks = (tid & 3) * 8;    // A: 64 rows x 32 k
    int bcp = tid * 8;                           // B: 2048 contiguous halves
    for (int k0 = 0; k0 < Ks; k0 += 32) {
        __syncthreads();
        {
            const _Float16* gp = A + (rowBase + arow) * (long)Ks + k0 + aks;
#pragma unroll
            for (int j = 0; j < 8; ++j) smA[arow * 40 + aks + j] = gp[j];
        }
        {
            const _Float16* gp = Bsw + ((long)(k0 >> 5) * NT + (colBase >> 4)) * 512 + bcp;
#pragma unroll
            for (int j = 0; j < 8; ++j) smB[bcp + j] = gp[j];
        }
        __syncthreads();
        wmma_step(smA, smB, mt, ntB, lane, l, hi, c0, c1);
    }
}

// 9-tap implicit-GEMM core for 3x3 convs.  A is a [T][Kp] f16 activation image
// (NHWC, padded channels), Bsw is tap-major swizzled weights (Ktot = 9*Kp).
// A 64-row block covers 64 consecutive pixels of one image row (256 % 64 == 0),
// so only the +-1 column shift needs per-thread validity; zero-fill handles pad.
__device__ __forceinline__ void conv_gemm_tiles(const _Float16* __restrict__ A,
                                                const _Float16* __restrict__ Bsw,
                                                long rowBase, int colBase, int Kp, int NT,
                                                _Float16* smA, _Float16* smB,
                                                v8f& c0, v8f& c1) {
    int tid = threadIdx.x, wave = tid >> 5, lane = tid & 31;
    int l = lane & 15, hi = lane >> 4;
    int mt = wave & 3, ntB = (wave >> 2) * 2;
    int arow = tid >> 2, aks = (tid & 3) * 8;
    int bcp = tid * 8;
    long p = rowBase + arow;
    int bb = (int)(p >> 16);
    int hw = (int)(p & 65535);
    int hh = hw >> 8, ww = hw & 255;
    for (int t9 = 0; t9 < 9; ++t9) {
        int py = hh + (t9 / 3) - 1;
        int px = ww + (t9 % 3) - 1;
        bool valid = ((unsigned)py < 256u) && ((unsigned)px < 256u);
        const _Float16* arp = A + ((long)bb * 65536 + py * 256 + px) * Kp;
        for (int k0 = 0; k0 < Kp; k0 += 32) {
            __syncthreads();
            if (valid) {
                const _Float16* gp = arp + k0 + aks;
#pragma unroll
                for (int j = 0; j < 8; ++j) smA[arow * 40 + aks + j] = gp[j];
            } else {
#pragma unroll
                for (int j = 0; j < 8; ++j) smA[arow * 40 + aks + j] = (_Float16)0.0f;
            }
            {
                int kb = (t9 * Kp + k0) >> 5;
                const _Float16* gp = Bsw + ((long)kb * NT + (colBase >> 4)) * 512 + bcp;
#pragma unroll
                for (int j = 0; j < 8; ++j) smB[bcp + j] = gp[j];
            }
            __syncthreads();
            wmma_step(smA, smB, mt, ntB, lane, l, hi, c0, c1);
        }
    }
}

// conv1: xn_h [T,192] (*) wc1 -> gelu -> y1_h [T,64] (pads zeroed inline)
__global__ __launch_bounds__(256) void conv1_gemm(const _Float16* __restrict__ A,
                                                  const _Float16* __restrict__ B,
                                                  const float* __restrict__ bias,
                                                  _Float16* __restrict__ y1_h) {
    __shared__ _Float16 smA[64 * 40];
    __shared__ _Float16 smB[2048];
    long rowBase = (long)blockIdx.x * 64;
    v8f c0 = {}, c1 = {};
    conv_gemm_tiles(A, B, rowBase, 0, DIMP, CDIMP / 16, smA, smB, c0, c1);
    int lane = threadIdx.x & 31, l = lane & 15, hi = lane >> 4;
    int wave = threadIdx.x >> 5, mt = wave & 3, ntB = (wave >> 2) * 2;
#pragma unroll
    for (int r = 0; r < 8; ++r) {
        long row = rowBase + mt * 16 + r + hi * 8;
#pragma unroll
        for (int i = 0; i < 2; ++i) {
            int col = (ntB + i) * 16 + l;
            float v = (i ? c1[r] : c0[r]);
            y1_h[row * CDIMP + col] =
                (col < CDIM) ? (_Float16)gelu_exact(v + bias[col]) : (_Float16)0.0f;
        }
    }
}

// conv2: y1_h [T,64] (*) wc2 -> y2 fp32 [T,180]
__global__ __launch_bounds__(256) void conv2_gemm(const _Float16* __restrict__ A,
                                                  const _Float16* __restrict__ B,
                                                  const float* __restrict__ bias,
                                                  float* __restrict__ y2) {
    __shared__ _Float16 smA[64 * 40];
    __shared__ _Float16 smB[2048];
    long rowBase = (long)blockIdx.x * 64;
    int colBase = blockIdx.y * 64;
    v8f c0 = {}, c1 = {};
    conv_gemm_tiles(A, B, rowBase, colBase, CDIMP, DIMP / 16, smA, smB, c0, c1);
    int lane = threadIdx.x & 31, l = lane & 15, hi = lane >> 4;
    int wave = threadIdx.x >> 5, mt = wave & 3, ntB = (wave >> 2) * 2;
#pragma unroll
    for (int r = 0; r < 8; ++r) {
        long row = rowBase + mt * 16 + r + hi * 8;
#pragma unroll
        for (int i = 0; i < 2; ++i) {
            int col = colBase + (ntB + i) * 16 + l;
            if (col < DIM) y2[row * DIM + col] = (i ? c1[r] : c0[r]) + bias[col];
        }
    }
}

// qkv GEMM: xn_h[T,192] @ wq + qkv_b -> qkv_h[t][which*192 + head*32 + d];
// the 36 dead padded columns are mapped onto the d=30/31 pad slots and zeroed.
__global__ __launch_bounds__(256) void gemm_qkv(const _Float16* __restrict__ A,
                                                const _Float16* __restrict__ B,
                                                const float* __restrict__ bias,
                                                _Float16* __restrict__ qkv_h) {
    __shared__ _Float16 smA[64 * 40];
    __shared__ _Float16 smB[2048];
    long rowBase = (long)blockIdx.x * 64;
    int colBase = blockIdx.y * 64;
    v8f c0 = {}, c1 = {};
    gemm_tiles(A, B, rowBase, colBase, DIMP, NS_QKV / 16, smA, smB, c0, c1);
    int lane = threadIdx.x & 31, l = lane & 15, hi = lane >> 4;
    int wave = threadIdx.x >> 5, mt = wave & 3, ntB = (wave >> 2) * 2;
#pragma unroll
    for (int r = 0; r < 8; ++r) {
        long row = rowBase + mt * 16 + r + hi * 8;
#pragma unroll
        for (int i = 0; i < 2; ++i) {
            int col = colBase + (ntB + i) * 16 + l;
            if (col < 3 * DIM) {
                float v = (i ? c1[r] : c0[r]) + bias[col];
                int which = col / DIM, rem = col % DIM, head = rem / HD, d = rem % HD;
                qkv_h[row * QKVP + which * DIMP + head * 32 + d] = (_Float16)v;
            } else {                      // col in [540,576): zero a pad slot
                int j = col - 3 * DIM;    // 0..35
                int which2 = j / 12, head2 = (j % 12) / 2, d2 = HD + (j & 1);
                qkv_h[row * QKVP + which2 * DIMP + head2 * 32 + d2] = (_Float16)0.0f;
            }
        }
    }
}

// proj GEMM + residual + conv-branch fuse: x2 = x + (attn @ Wp + b) + 0.01*y2*ca
__global__ __launch_bounds__(256) void gemm_proj(const _Float16* __restrict__ A,
                                                 const _Float16* __restrict__ B,
                                                 const float* __restrict__ bias,
                                                 const float* __restrict__ x_in,
                                                 const float* __restrict__ y2,
                                                 const float* __restrict__ ca_a,
                                                 float* __restrict__ x2) {
    __shared__ _Float16 smA[64 * 40];
    __shared__ _Float16 smB[2048];
    long rowBase = (long)blockIdx.x * 64;
    int colBase = blockIdx.y * 64;
    v8f c0 = {}, c1 = {};
    gemm_tiles(A, B, rowBase, colBase, DIMP, DIMP / 16, smA, smB, c0, c1);
    int lane = threadIdx.x & 31, l = lane & 15, hi = lane >> 4;
    int wave = threadIdx.x >> 5, mt = wave & 3, ntB = (wave >> 2) * 2;
#pragma unroll
    for (int r = 0; r < 8; ++r) {
        long row = rowBase + mt * 16 + r + hi * 8;
        int bb = (int)(row >> 16);
#pragma unroll
        for (int i = 0; i < 2; ++i) {
            int col = colBase + (ntB + i) * 16 + l;
            if (col < DIM) {
                float v = (i ? c1[r] : c0[r]) + bias[col];
                float conv = y2[row * DIM + col] * ca_a[bb * DIM + col];
                x2[row * DIM + col] = x_in[row * DIM + col] + v + 0.01f * conv;
            }
        }
    }
}

// fc1 GEMM + exact GELU -> hid_h [T,736] f16 (pads zeroed inline)
__global__ __launch_bounds__(256) void gemm_fc1(const _Float16* __restrict__ A,
                                                const _Float16* __restrict__ B,
                                                const float* __restrict__ bias,
                                                _Float16* __restrict__ hid_h) {
    __shared__ _Float16 smA[64 * 40];
    __shared__ _Float16 smB[2048];
    long rowBase = (long)blockIdx.x * 64;
    int colBase = blockIdx.y * 64;
    v8f c0 = {}, c1 = {};
    gemm_tiles(A, B, rowBase, colBase, DIMP, NS_FC1 / 16, smA, smB, c0, c1);
    int lane = threadIdx.x & 31, l = lane & 15, hi = lane >> 4;
    int wave = threadIdx.x >> 5, mt = wave & 3, ntB = (wave >> 2) * 2;
#pragma unroll
    for (int r = 0; r < 8; ++r) {
        long row = rowBase + mt * 16 + r + hi * 8;
#pragma unroll
        for (int i = 0; i < 2; ++i) {
            int col = colBase + (ntB + i) * 16 + l;
            if (col < HID) {
                float v = gelu_exact((i ? c1[r] : c0[r]) + bias[col]);
                hid_h[row * HIDP + col] = (_Float16)v;
            } else if (col < HIDP) {
                hid_h[row * HIDP + col] = (_Float16)0.0f;
            }
        }
    }
}

// fc2 GEMM + final residual -> d_out (fp32, [2,65536,180])
__global__ __launch_bounds__(256) void gemm_fc2(const _Float16* __restrict__ A,
                                                const _Float16* __restrict__ B,
                                                const float* __restrict__ bias,
                                                const float* __restrict__ x2,
                                                float* __restrict__ out) {
    __shared__ _Float16 smA[64 * 40];
    __shared__ _Float16 smB[2048];
    long rowBase = (long)blockIdx.x * 64;
    int colBase = blockIdx.y * 64;
    v8f c0 = {}, c1 = {};
    gemm_tiles(A, B, rowBase, colBase, HIDP, DIMP / 16, smA, smB, c0, c1);
    int lane = threadIdx.x & 31, l = lane & 15, hi = lane >> 4;
    int wave = threadIdx.x >> 5, mt = wave & 3, ntB = (wave >> 2) * 2;
#pragma unroll
    for (int r = 0; r < 8; ++r) {
        long row = rowBase + mt * 16 + r + hi * 8;
#pragma unroll
        for (int i = 0; i < 2; ++i) {
            int col = colBase + (ntB + i) * 16 + l;
            if (col < DIM) {
                float v = (i ? c1[r] : c0[r]) + bias[col];
                out[row * DIM + col] = x2[row * DIM + col] + v;
            }
        }
    }
}

// ---------------------------------------------------------------------------
// Shifted-window attention. One block = (window, head, 64-query block).
// S kept in 8 accumulators/wave; RPB cached in LDS; mask computed arithmetically;
// two-pass softmax with wave32 shuffles; P@V via WMMA from LDS.
// ---------------------------------------------------------------------------
__global__ __launch_bounds__(256) void attn_kernel(const _Float16* __restrict__ qkv_h,
                                                   const float* __restrict__ rpb_table,
                                                   _Float16* __restrict__ attn_h) {
    __shared__ _Float16 smQ[64 * 36];      // Q rows (stride 36)
    __shared__ _Float16 smKV[256 * 36];    // K rows (stride 36) then Vt [32][264]
    __shared__ _Float16 smP[64 * 264];     // softmaxed P, f16, stride 264
    __shared__ float smRpb[961];           // rel-pos bias for this head
    __shared__ float redmax[2][64];
    __shared__ float redsum[2][64];

    int tid = threadIdx.x, wave = tid >> 5, lane = tid & 31;
    int l = lane & 15, hi = lane >> 4;
    int bid = blockIdx.x;
    int rb = bid & 3;                 // 64-row block within window
    int head = (bid >> 2) % NH;
    int wid = bid / (4 * NH);         // 0..511
    int b = wid >> 8, wrem = wid & 255, wi = wrem >> 4, wj = wrem & 15;

    for (int i = tid; i < 961; i += 256) smRpb[i] = rpb_table[i * 6 + head];

    // load K (all 256 key tokens of this window, this head)
    {
        int j = tid;
        int hr = wi * WSZ + (j >> 4), wr = wj * WSZ + (j & 15);
        long t = (long)b * 65536 + (long)((hr + SHIFT) & 255) * 256 + ((wr + SHIFT) & 255);
        const _Float16* src = qkv_h + t * QKVP + DIMP + head * 32;   // which=1 (K)
#pragma unroll
        for (int d = 0; d < 32; ++d) smKV[j * 36 + d] = src[d];
    }
    // load Q (64 query rows of this block)
    if (tid < 64) {
        int i = rb * 64 + tid;
        int hr = wi * WSZ + (i >> 4), wr = wj * WSZ + (i & 15);
        long t = (long)b * 65536 + (long)((hr + SHIFT) & 255) * 256 + ((wr + SHIFT) & 255);
        const _Float16* src = qkv_h + t * QKVP + head * 32;          // which=0 (Q)
#pragma unroll
        for (int d = 0; d < 32; ++d) smQ[tid * 36 + d] = src[d];
    }
    __syncthreads();

    int mt = wave & 3;          // 16-row tile within 64
    int ntg = wave >> 2;        // key half: 0 -> cols 0..127, 1 -> cols 128..255

    // ---- GEMM1: S = Q @ K^T (K=32, single WMMA per 16x16 tile, 8 tiles/wave) ----
    v16h a;
    {
        const _Float16* p = smQ + (mt * 16 + l) * 36;
#pragma unroll
        for (int h = 0; h < 16; ++h) a[h] = p[kmapA(h, hi)];
    }
    v8f acc[8] = {};
#pragma unroll
    for (int i = 0; i < 8; ++i) {
        int n0 = ntg * 128 + i * 16;
        v16h bf;
        {
            const _Float16* p = smKV + (n0 + l) * 36 + hi * 16;
#pragma unroll
            for (int h = 0; h < 16; ++h) bf[h] = p[h];
        }
        acc[i] = __builtin_amdgcn_wmma_f32_16x16x32_f16(false, a, false, bf, (short)0, acc[i], false, false);
    }

    // ---- scale + RPB + shift-mask, then softmax over 256 keys ----
    const float scale = 0.18257418583505536f;   // 30^-0.5
    float rstat[8];
#pragma unroll
    for (int r = 0; r < 8; ++r) {
        int rowl = mt * 16 + r + hi * 8;
        int irow = rb * 64 + rowl;
        int yi = irow >> 4, xi = irow & 15;
        int gri = grp(wi * WSZ + yi) * 3 + grp(wj * WSZ + xi);
        float mx = -3.0e38f;
#pragma unroll
        for (int i = 0; i < 8; ++i) {
            int j = ntg * 128 + i * 16 + l;
            int yj = j >> 4, xj = j & 15;
            int grj = grp(wi * WSZ + yj) * 3 + grp(wj * WSZ + xj);
            float v = acc[i][r] * scale + smRpb[(yi - yj + 15) * 31 + (xi - xj + 15)]
                      + ((gri != grj) ? -100.0f : 0.0f);
            acc[i][r] = v;
            mx = fmaxf(mx, v);
        }
#pragma unroll
        for (int m = 8; m; m >>= 1) mx = fmaxf(mx, __shfl_xor(mx, m, 32));
        rstat[r] = mx;
    }
    if (l == 0) {
#pragma unroll
        for (int r = 0; r < 8; ++r) redmax[ntg][mt * 16 + r + hi * 8] = rstat[r];
    }
    __syncthreads();
#pragma unroll
    for (int r = 0; r < 8; ++r) {
        int rowl = mt * 16 + r + hi * 8;
        float mx = fmaxf(redmax[0][rowl], redmax[1][rowl]);
        float s = 0.f;
#pragma unroll
        for (int i = 0; i < 8; ++i) { acc[i][r] = expf(acc[i][r] - mx); s += acc[i][r]; }
#pragma unroll
        for (int m = 8; m; m >>= 1) s += __shfl_xor(s, m, 32);
        rstat[r] = s;
    }
    if (l == 0) {
#pragma unroll
        for (int r = 0; r < 8; ++r) redsum[ntg][mt * 16 + r + hi * 8] = rstat[r];
    }
    __syncthreads();
#pragma unroll
    for (int r = 0; r < 8; ++r) {
        int rowl = mt * 16 + r + hi * 8;
        float inv = 1.0f / (redsum[0][rowl] + redsum[1][rowl]);
#pragma unroll
        for (int i = 0; i < 8; ++i)
            smP[rowl * 264 + ntg * 128 + i * 16 + l] = (_Float16)(acc[i][r] * inv);
    }
    __syncthreads();

    // ---- load V transposed into smKV: Vt[d][j], stride 264 ----
    {
        int j = tid;
        int hr = wi * WSZ + (j >> 4), wr = wj * WSZ + (j & 15);
        long t = (long)b * 65536 + (long)((hr + SHIFT) & 255) * 256 + ((wr + SHIFT) & 255);
        const _Float16* src = qkv_h + t * QKVP + 2 * DIMP + head * 32;   // which=2 (V)
#pragma unroll
        for (int d = 0; d < 32; ++d) smKV[d * 264 + j] = src[d];
    }
    __syncthreads();

    // ---- GEMM2: O = P @ V, 64x32 output, 8 tiles total (one per wave) ----
    int nt2 = wave >> 2;        // d tile: 0 -> d 0..15, 1 -> d 16..31
    v8f o = {};
#pragma unroll
    for (int ks = 0; ks < 8; ++ks) {
        v16h af, bf;
        {
            const _Float16* p = smP + (mt * 16 + l) * 264 + ks * 32;
#pragma unroll
            for (int h = 0; h < 16; ++h) af[h] = p[kmapA(h, hi)];
        }
        {
            const _Float16* p = smKV + (nt2 * 16 + l) * 264 + ks * 32 + hi * 16;
#pragma unroll
            for (int h = 0; h < 16; ++h) bf[h] = p[h];
        }
        o = __builtin_amdgcn_wmma_f32_16x16x32_f16(false, af, false, bf, (short)0, o, false, false);
    }
#pragma unroll
    for (int r = 0; r < 8; ++r) {
        int rowl = mt * 16 + r + hi * 8;
        int irow = rb * 64 + rowl;
        int hr = wi * WSZ + (irow >> 4), wr = wj * WSZ + (irow & 15);
        long t = (long)b * 65536 + (long)((hr + SHIFT) & 255) * 256 + ((wr + SHIFT) & 255);
        int d = nt2 * 16 + l;
        if (d < HD) attn_h[t * DIMP + head * HD + d] = (_Float16)o[r];
    }
}

// ---------------------------------------------------------------------------
// Host-side orchestration (graph-capture safe: only async kernel launches).
// ---------------------------------------------------------------------------
extern "C" void kernel_launch(void* const* d_in, const int* in_sizes, int n_in,
                              void* d_out, int out_size, void* d_ws, size_t ws_size,
                              hipStream_t stream) {
    const float* x       = (const float*)d_in[0];
    // d_in[1] = h (256), d_in[2] = w (256): fixed, hardcoded
    const float* norm1_g = (const float*)d_in[3];
    const float* norm1_b = (const float*)d_in[4];
    const float* qkv_w   = (const float*)d_in[5];
    const float* qkv_b   = (const float*)d_in[6];
    const float* rpb_tab = (const float*)d_in[7];
    const float* proj_w  = (const float*)d_in[8];
    const float* proj_b  = (const float*)d_in[9];
    const float* cab_w1  = (const float*)d_in[10];
    const float* cab_b1  = (const float*)d_in[11];
    const float* cab_w2  = (const float*)d_in[12];
    const float* cab_b2  = (const float*)d_in[13];
    const float* ca_w1   = (const float*)d_in[14];
    const float* ca_b1   = (const float*)d_in[15];
    const float* ca_w2   = (const float*)d_in[16];
    const float* ca_b2   = (const float*)d_in[17];
    const float* norm2_g = (const float*)d_in[18];
    const float* norm2_b = (const float*)d_in[19];
    const float* fc1_w   = (const float*)d_in[20];
    const float* fc1_b   = (const float*)d_in[21];
    const float* fc2_w   = (const float*)d_in[22];
    const float* fc2_b   = (const float*)d_in[23];

    // workspace carve-out (256B aligned)
    size_t off = 0;
    auto carve = [&](size_t bytes) -> void* {
        off = (off + 255) & ~(size_t)255;
        void* p = (char*)d_ws + off;
        off += bytes;
        return p;
    };
    float*     y2    = (float*)    carve((size_t)T_TOK * DIM * 4);
    float*     x2    = (float*)    carve((size_t)T_TOK * DIM * 4);
    float*     pool  = (float*)    carve(2 * DIM * 4);
    float*     ca_a  = (float*)    carve(2 * DIM * 4);
    _Float16*  xn_h  = (_Float16*) carve((size_t)T_TOK * DIMP * 2);
    _Float16*  x2n_h = (_Float16*) carve((size_t)T_TOK * DIMP * 2);
    _Float16*  qkv_h = (_Float16*) carve((size_t)T_TOK * QKVP * 2);
    _Float16*  att_h = (_Float16*) carve((size_t)T_TOK * DIMP * 2);
    _Float16*  hid_h = (_Float16*) carve((size_t)T_TOK * HIDP * 2);
    _Float16*  y1_h  = (_Float16*) carve((size_t)T_TOK * CDIMP * 2);
    _Float16*  wq    = (_Float16*) carve((size_t)DIMP * NS_QKV * 2);
    _Float16*  wp    = (_Float16*) carve((size_t)DIMP * DIMP * 2);
    _Float16*  w1    = (_Float16*) carve((size_t)DIMP * NS_FC1 * 2);
    _Float16*  w2    = (_Float16*) carve((size_t)HIDP * DIMP * 2);
    _Float16*  wc1   = (_Float16*) carve((size_t)9 * DIMP * CDIMP * 2);
    _Float16*  wc2   = (_Float16*) carve((size_t)9 * CDIMP * DIMP * 2);
    (void)ws_size; (void)in_sizes; (void)n_in; (void)out_size;

    // zero pad columns (cols 180..191) of 192-stride f16 buffers
    zero_pads<<<(T_TOK * 12) / 256, 256, 0, stream>>>(xn_h);
    zero_pads<<<(T_TOK * 12) / 256, 256, 0, stream>>>(x2n_h);
    zero_pads<<<(T_TOK * 12) / 256, 256, 0, stream>>>(att_h);

    // weight conversion (fp32 -> f16, WMMA-fragment-swizzled B operands)
    convert_wT<<<(DIMP * NS_QKV + 255) / 256, 256, 0, stream>>>(qkv_w,  wq, 3 * DIM, DIM, NS_QKV, DIMP);
    convert_wT<<<(DIMP * DIMP   + 255) / 256, 256, 0, stream>>>(proj_w, wp, DIM,     DIM, DIMP,   DIMP);
    convert_wT<<<(DIMP * NS_FC1 + 255) / 256, 256, 0, stream>>>(fc1_w,  w1, HID,     DIM, NS_FC1, DIMP);
    convert_wT<<<(HIDP * DIMP   + 255) / 256, 256, 0, stream>>>(fc2_w,  w2, DIM,     HID, DIMP,   HIDP);
    convert_conv_w<<<(9 * DIMP * CDIMP + 255) / 256, 256, 0, stream>>>(cab_w1, wc1, CDIM, DIM, DIMP, CDIMP);
    convert_conv_w<<<(9 * CDIMP * DIMP + 255) / 256, 256, 0, stream>>>(cab_w2, wc2, DIM, CDIM, CDIMP, DIMP);

    // LN1 -> xn_h (f16 for qkv GEMM + conv implicit GEMM)
    ln_kernel<<<T_TOK / 8, 256, 0, stream>>>(x, norm1_g, norm1_b, xn_h);

    // CAB conv branch (implicit GEMM on WMMA, activations L2-resident)
    conv1_gemm<<<dim3(T_TOK / 64, 1), 256, 0, stream>>>(xn_h, wc1, cab_b1, y1_h);
    conv2_gemm<<<dim3(T_TOK / 64, DIMP / 64), 256, 0, stream>>>(y1_h, wc2, cab_b2, y2);
    pool_kernel<<<2 * DIM, 256, 0, stream>>>(y2, pool);
    ca_kernel<<<2, 192, 0, stream>>>(pool, ca_w1, ca_b1, ca_w2, ca_b2, ca_a);

    // attention path
    gemm_qkv<<<dim3(T_TOK / 64, NS_QKV / 64), 256, 0, stream>>>(xn_h, wq, qkv_b, qkv_h);
    attn_kernel<<<512 * NH * 4, 256, 0, stream>>>(qkv_h, rpb_tab, att_h);
    gemm_proj<<<dim3(T_TOK / 64, DIMP / 64), 256, 0, stream>>>(att_h, wp, proj_b, x, y2, ca_a, x2);

    // MLP path
    ln_kernel<<<T_TOK / 8, 256, 0, stream>>>(x2, norm2_g, norm2_b, x2n_h);
    gemm_fc1<<<dim3(T_TOK / 64, NS_FC1 / 64), 256, 0, stream>>>(x2n_h, w1, fc1_b, hid_h);
    gemm_fc2<<<dim3(T_TOK / 64, DIMP / 64), 256, 0, stream>>>(hid_h, w2, fc2_b, x2, (float*)d_out);
}
